// MultiHeadAttention_77283641524724
// MI455X (gfx1250) — compile-verified, matrix operand order checked by reading, not run on hardware
//
#include <hip/hip_runtime.h>
#include <hip/hip_bf16.h>

// Problem constants (match reference)
#define BB 2
#define SS 2048
#define EE 1024
#define HH 16
#define DD 64
#define NROWS (BB * SS)          // 4096
#define NX (BB * SS * EE)        // 4,194,304
#define NW (HH * EE * DD)        // 1,048,576
#define NBHS (BB * HH * SS)      // 65,536
#define LOG2E 1.44269504088896f
#define SCALE_ 0.03125f          // 1/sqrt(1024)

typedef __attribute__((ext_vector_type(16))) __bf16 v16bf;
typedef __attribute__((ext_vector_type(8)))  float  v8f;

union FragU { v16bf v; uint4 q[2]; };

__device__ __forceinline__ unsigned short f2bf(float f) {
  unsigned int u = __float_as_uint(f);
  u += 0x7FFFu + ((u >> 16) & 1u);         // round-to-nearest-even
  return (unsigned short)(u >> 16);
}
__device__ __forceinline__ float bf2f_lo(unsigned int u) {
  return __uint_as_float(u << 16);
}
__device__ __forceinline__ float bf2f_hi(unsigned int u) {
  return __uint_as_float(u & 0xFFFF0000u);
}

// Fragment loader for 16x32 A (row = M) or 32x16 B stored N-major [N][K].
// ISA 16-bit layout: lane<16 -> row=lane, K elems {0..7},{16..23};
//                    lane>=16 -> row=lane-16, K elems {8..15},{24..31}.
__device__ __forceinline__ v16bf load_frag(const unsigned short* base, int rowStride) {
  int lane = threadIdx.x & 31;
  int r = lane & 15;
  int sel = lane >> 4;
  const unsigned short* p = base + r * rowStride + 8 * sel;
  FragU u;
  u.q[0] = *(const uint4*)(p);
  u.q[1] = *(const uint4*)(p + 16);
  return u.v;
}

__device__ __forceinline__ v8f wmma_bf16(v16bf a, v16bf b, v8f c) {
  return __builtin_amdgcn_wmma_f32_16x16x32_bf16(false, a, false, b, (short)0, c,
                                                 false, false);
}

// ---------------------------------------------------------------- conversions
// 8 elements per thread: 2x float4 in, 1x uint4 (8 bf16) out, for q/k/v at once.
__global__ __launch_bounds__(256) void cvt_x_kernel(const float* __restrict__ q,
                                                    const float* __restrict__ k,
                                                    const float* __restrict__ v,
                                                    unsigned short* __restrict__ xq,
                                                    unsigned short* __restrict__ xk,
                                                    unsigned short* __restrict__ xv) {
  int i = blockIdx.x * 256 + threadIdx.x;   // over NX/8 groups
  long base = (long)i * 8;
  const float4* q4 = (const float4*)(q + base);
  const float4* k4 = (const float4*)(k + base);
  const float4* v4 = (const float4*)(v + base);
  float4 a0 = q4[0], a1 = q4[1];
  float4 b0 = k4[0], b1 = k4[1];
  float4 c0 = v4[0], c1 = v4[1];
  uint4 pq, pk, pv;
  pq.x = f2bf(a0.x) | ((unsigned)f2bf(a0.y) << 16);
  pq.y = f2bf(a0.z) | ((unsigned)f2bf(a0.w) << 16);
  pq.z = f2bf(a1.x) | ((unsigned)f2bf(a1.y) << 16);
  pq.w = f2bf(a1.z) | ((unsigned)f2bf(a1.w) << 16);
  pk.x = f2bf(b0.x) | ((unsigned)f2bf(b0.y) << 16);
  pk.y = f2bf(b0.z) | ((unsigned)f2bf(b0.w) << 16);
  pk.z = f2bf(b1.x) | ((unsigned)f2bf(b1.y) << 16);
  pk.w = f2bf(b1.z) | ((unsigned)f2bf(b1.w) << 16);
  pv.x = f2bf(c0.x) | ((unsigned)f2bf(c0.y) << 16);
  pv.y = f2bf(c0.z) | ((unsigned)f2bf(c0.w) << 16);
  pv.z = f2bf(c1.x) | ((unsigned)f2bf(c1.y) << 16);
  pv.w = f2bf(c1.z) | ((unsigned)f2bf(c1.w) << 16);
  *(uint4*)(xq + base) = pq;
  *(uint4*)(xk + base) = pk;
  *(uint4*)(xv + base) = pv;
}

// W[h][e][d] (f32) -> WT[h][d][e] (bf16), for all three weights in one pass.
__global__ __launch_bounds__(256) void cvt_w_kernel(const float* __restrict__ wq,
                                                    const float* __restrict__ wk,
                                                    const float* __restrict__ wv,
                                                    unsigned short* __restrict__ wqt,
                                                    unsigned short* __restrict__ wkt,
                                                    unsigned short* __restrict__ wvt) {
  int i = blockIdx.x * 256 + threadIdx.x;
  if (i >= NW) return;
  int h = i >> 16;            // / (EE*DD)
  int rem = i & 65535;
  int e = rem >> 6;           // / DD
  int d = rem & 63;
  int dst = h * (DD * EE) + d * EE + e;
  wqt[dst] = f2bf(wq[i]);
  wkt[dst] = f2bf(wk[i]);
  wvt[dst] = f2bf(wv[i]);
}

// ---------------------------------------------------- per-head projection GEMM
// One wave: C[32 x 64] = X[32 x 1024] * W_h[1024 x 64] + bias.
// Software-pipelined (double-buffered fragments, 2 K-steps per iteration).
// mode 0/1: store bf16 row-major [B,H,S,64]; mode 2: store bf16 transposed [B,H,64,S]
__global__ __launch_bounds__(128) void proj_gemm_kernel(const unsigned short* __restrict__ X,
                                                        const unsigned short* __restrict__ WT,
                                                        const float* __restrict__ bias,
                                                        unsigned short* __restrict__ outp,
                                                        int mode) {
  int wave = threadIdx.x >> 5;
  int tile = blockIdx.x * 4 + wave;          // 2048 tiles = 128 rowTiles * 16 heads
  int h = tile >> 7;
  int rowTile = tile & 127;
  int m0 = rowTile * 32;                     // row in [0, B*S)
  int lane = threadIdx.x & 31;
  int r = lane & 15;
  int hi = lane >> 4;

  v8f acc[2][4] = {};
  const unsigned short* wbase = WT + h * (DD * EE);
  const unsigned short* xbase = X + (long)m0 * EE;

  v16bf a0[2], b0[4], a1[2], b1[4];
#pragma unroll
  for (int m = 0; m < 2; ++m) a0[m] = load_frag(xbase + m * 16 * EE, EE);
#pragma unroll
  for (int f = 0; f < 4; ++f) b0[f] = load_frag(wbase + f * 16 * EE, EE);

  for (int e0 = 0; e0 < EE; e0 += 64) {
    // issue loads for e0+32 before consuming e0
#pragma unroll
    for (int m = 0; m < 2; ++m) a1[m] = load_frag(xbase + m * 16 * EE + e0 + 32, EE);
#pragma unroll
    for (int f = 0; f < 4; ++f) b1[f] = load_frag(wbase + f * 16 * EE + e0 + 32, EE);
#pragma unroll
    for (int m = 0; m < 2; ++m)
#pragma unroll
      for (int f = 0; f < 4; ++f) acc[m][f] = wmma_bf16(a0[m], b0[f], acc[m][f]);

    if (e0 + 64 < EE) {
#pragma unroll
      for (int m = 0; m < 2; ++m) a0[m] = load_frag(xbase + m * 16 * EE + e0 + 64, EE);
#pragma unroll
      for (int f = 0; f < 4; ++f) b0[f] = load_frag(wbase + f * 16 * EE + e0 + 64, EE);
    }
#pragma unroll
    for (int m = 0; m < 2; ++m)
#pragma unroll
      for (int f = 0; f < 4; ++f) acc[m][f] = wmma_bf16(a1[m], b1[f], acc[m][f]);
  }

  int b = m0 >> 11;                          // / SS
  int s0 = m0 & (SS - 1);
#pragma unroll
  for (int m = 0; m < 2; ++m)
#pragma unroll
    for (int f = 0; f < 4; ++f) {
      float bv = bias[h * DD + f * 16 + r];
#pragma unroll
      for (int i = 0; i < 8; ++i) {
        float val = acc[m][f][i] + bv;
        int srow = s0 + m * 16 + i + 8 * hi;
        if (mode == 2) {
          outp[((long)((b * HH + h) * DD + f * 16 + r)) * SS + srow] = f2bf(val);
        } else {
          outp[((long)((b * HH + h) * SS + srow)) * DD + f * 16 + r] = f2bf(val);
        }
      }
    }
}

// ------------------------------------------------------------ row squared norms
__global__ __launch_bounds__(256) void rownorm_kernel(const unsigned short* __restrict__ Qb,
                                                      const unsigned short* __restrict__ Kb,
                                                      float* __restrict__ qn,
                                                      float* __restrict__ kn) {
  int i = blockIdx.x * 256 + threadIdx.x;    // over B*H*S
  if (i >= NBHS) return;
  const uint4* pq = (const uint4*)(Qb + (long)i * DD);
  const uint4* pk = (const uint4*)(Kb + (long)i * DD);
  float sq = 0.f, sk = 0.f;
#pragma unroll
  for (int j = 0; j < 8; ++j) {
    uint4 a = pq[j];
    uint4 b = pk[j];
    float x;
    x = bf2f_lo(a.x); sq += x * x; x = bf2f_hi(a.x); sq += x * x;
    x = bf2f_lo(a.y); sq += x * x; x = bf2f_hi(a.y); sq += x * x;
    x = bf2f_lo(a.z); sq += x * x; x = bf2f_hi(a.z); sq += x * x;
    x = bf2f_lo(a.w); sq += x * x; x = bf2f_hi(a.w); sq += x * x;
    x = bf2f_lo(b.x); sk += x * x; x = bf2f_hi(b.x); sk += x * x;
    x = bf2f_lo(b.y); sk += x * x; x = bf2f_hi(b.y); sk += x * x;
    x = bf2f_lo(b.z); sk += x * x; x = bf2f_hi(b.z); sk += x * x;
    x = bf2f_lo(b.w); sk += x * x; x = bf2f_hi(b.w); sk += x * x;
  }
  qn[i] = sq;
  kn[i] = sk;
}

// ------------------------------------------------------- flash attention kernel
// One wave = 32 queries of one (b,h): every lane owns a softmax row (no
// divergence). Keys streamed in blocks of 32 with online softmax; next block's
// K fragments and this block's V fragments are issued before the softmax chain
// so exp/LDS work hides their latency.
__global__ __launch_bounds__(128) void flash_attn_kernel(const unsigned short* __restrict__ Qb,
                                                         const unsigned short* __restrict__ Kb,
                                                         const unsigned short* __restrict__ Vt,
                                                         const float* __restrict__ qn,
                                                         const float* __restrict__ kn,
                                                         float* __restrict__ aout) {
  __shared__ __align__(16) float sS[4][32 * 32];           // score tile per wave
  __shared__ __align__(16) unsigned short sP[4][32 * 32];  // P tile (bf16) per wave
  __shared__ float sAux[4][32];                            // per-row corr / l

  int wave = threadIdx.x >> 5;
  int lane = threadIdx.x & 31;
  int tile = blockIdx.x * 4 + wave;          // B*H*(S/32) = 2048 tiles
  int bh = tile >> 6;
  int qt = tile & 63;
  int s0 = qt * 32;
  int r = lane & 15;
  int hi = lane >> 4;

  float* mySS = sS[wave];
  unsigned short* myP = sP[wave];
  unsigned int* myP32 = (unsigned int*)myP;
  float* myAux = sAux[wave];

  // Q A-fragments: 2 M-subtiles x 2 dk-chunks, resident for the whole loop.
  const unsigned short* qbase = Qb + ((long)bh * SS + s0) * DD;
  v16bf qa[2][2];
#pragma unroll
  for (int m = 0; m < 2; ++m) {
    qa[m][0] = load_frag(qbase + m * 16 * DD, DD);
    qa[m][1] = load_frag(qbase + m * 16 * DD + 32, DD);
  }

  // -0.5*||q||^2 for the 8 C-rows per M-subtile owned by this lane.
  float qnh[2][8];
#pragma unroll
  for (int m = 0; m < 2; ++m) {
    const float* qp = qn + (long)bh * SS + s0 + m * 16 + 8 * hi;
#pragma unroll
    for (int i = 0; i < 8; ++i) qnh[m][i] = -0.5f * qp[i];
  }

  float m_prev = -1e30f;
  float lsum = 0.f;
  v8f o[2][4] = {};

  const unsigned short* kbh = Kb + (long)bh * SS * DD;
  const float* knrow = kn + (long)bh * SS;

  v16bf kbufA[4], kbufB[4];
  // preload K fragments for block 0: [t*2+c] = keys tile t, dk chunk c
#pragma unroll
  for (int t = 0; t < 2; ++t) {
    kbufA[t * 2 + 0] = load_frag(kbh + t * 16 * DD, DD);
    kbufA[t * 2 + 1] = load_frag(kbh + t * 16 * DD + 32, DD);
  }

  auto block = [&](int j0, v16bf* kc, v16bf* knext) {
    // --- scores: 8 WMMAs (2 M-subtiles x 2 key tiles x 2 dk chunks)
    float knv[2];
#pragma unroll
    for (int t = 0; t < 2; ++t) knv[t] = -0.5f * knrow[j0 + t * 16 + r];
#pragma unroll
    for (int m = 0; m < 2; ++m)
#pragma unroll
      for (int t = 0; t < 2; ++t) {
        v8f z = {};
        v8f sc = wmma_bf16(qa[m][0], kc[t * 2 + 0], z);
        sc = wmma_bf16(qa[m][1], kc[t * 2 + 1], sc);
#pragma unroll
        for (int i = 0; i < 8; ++i) {
          float val = (sc[i] + qnh[m][i] + knv[t]) * SCALE_;
          mySS[(m * 16 + i + 8 * hi) * 32 + t * 16 + r] = val;
        }
      }

    // --- issue V fragment loads for this block (consumed after softmax)
    v16bf vb[4];
#pragma unroll
    for (int f = 0; f < 4; ++f)
      vb[f] = load_frag(Vt + ((long)bh * DD + f * 16) * SS + j0, SS);

    // --- issue next block's K fragment loads (consumed next call)
    if (j0 + 32 < SS) {
#pragma unroll
      for (int t = 0; t < 2; ++t) {
        knext[t * 2 + 0] = load_frag(kbh + (j0 + 32 + t * 16) * DD, DD);
        knext[t * 2 + 1] = load_frag(kbh + (j0 + 32 + t * 16) * DD + 32, DD);
      }
      __builtin_prefetch(Vt + ((long)bh * DD) * SS + j0 + 32, 0, 1);
    }

    // --- online softmax: every lane owns one of the 32 query rows
    {
      float rowv[32];
      float tmax = -1e30f;
#pragma unroll
      for (int k = 0; k < 32; ++k) {
        rowv[k] = mySS[lane * 32 + k];
        tmax = fmaxf(tmax, rowv[k]);
      }
      float m_new = fmaxf(m_prev, tmax);
      float corr = __builtin_amdgcn_exp2f((m_prev - m_new) * LOG2E);
      float ssum = 0.f;
#pragma unroll
      for (int k = 0; k < 16; ++k) {
        float p0 = __builtin_amdgcn_exp2f((rowv[2 * k] - m_new) * LOG2E);
        float p1 = __builtin_amdgcn_exp2f((rowv[2 * k + 1] - m_new) * LOG2E);
        ssum += p0 + p1;
        myP32[lane * 16 + k] = (unsigned)f2bf(p0) | ((unsigned)f2bf(p1) << 16);
      }
      lsum = lsum * corr + ssum;
      m_prev = m_new;
      myAux[lane] = corr;
    }

    // --- rescale O accumulators by per-row correction
#pragma unroll
    for (int m = 0; m < 2; ++m) {
      float c8[8];
#pragma unroll
      for (int i = 0; i < 8; ++i) c8[i] = myAux[m * 16 + 8 * hi + i];
#pragma unroll
      for (int f = 0; f < 4; ++f)
#pragma unroll
        for (int i = 0; i < 8; ++i) o[m][f][i] *= c8[i];
    }

    // --- O += P(32x32) * V(32x64): 8 WMMAs; V loads were issued pre-softmax
    v16bf pa[2];
#pragma unroll
    for (int m = 0; m < 2; ++m) pa[m] = load_frag(myP + m * 16 * 32, 32);
#pragma unroll
    for (int m = 0; m < 2; ++m)
#pragma unroll
      for (int f = 0; f < 4; ++f) o[m][f] = wmma_bf16(pa[m], vb[f], o[m][f]);
  };

  for (int j0 = 0; j0 < SS; j0 += 64) {
    block(j0, kbufA, kbufB);
    block(j0 + 32, kbufB, kbufA);
  }

  // --- finalize: divide by l, write concat-head layout [B,S,E] (f32)
  myAux[lane] = (lsum > 1e-30f) ? lsum : 1e-30f;
  int b = bh >> 4;
  int h = bh & 15;
#pragma unroll
  for (int m = 0; m < 2; ++m) {
    float l8[8];
#pragma unroll
    for (int i = 0; i < 8; ++i) l8[i] = myAux[m * 16 + 8 * hi + i];
#pragma unroll
    for (int f = 0; f < 4; ++f)
#pragma unroll
      for (int i = 0; i < 8; ++i) {
        int srow = s0 + m * 16 + i + 8 * hi;
        aout[((long)(b * SS + srow)) * EE + h * DD + f * 16 + r] = o[m][f][i] / l8[i];
      }
  }
}

// ------------------------------------------------- residual + custom LayerNorm
__global__ __launch_bounds__(256) void resid_ln_kernel(const float* __restrict__ attn,
                                                       const float* __restrict__ resid,
                                                       const float* __restrict__ gscale,
                                                       const float* __restrict__ gshift,
                                                       float* __restrict__ out) {
  int row = blockIdx.x;                      // B*S rows
  const float* a = attn + (long)row * EE;
  const float* rx = resid + (long)row * EE;
  float x[4];
  float lsum = 0.f, lsq = 0.f;
#pragma unroll
  for (int t = 0; t < 4; ++t) {
    int e = threadIdx.x + t * 256;
    float v = a[e] + rx[e];
    x[t] = v;
    lsum += v;
    lsq += v * v;
  }
  __shared__ float s1[256], s2[256];
  s1[threadIdx.x] = lsum;
  s2[threadIdx.x] = lsq;
  __syncthreads();
  for (int offs = 128; offs > 0; offs >>= 1) {
    if (threadIdx.x < offs) {
      s1[threadIdx.x] += s1[threadIdx.x + offs];
      s2[threadIdx.x] += s2[threadIdx.x + offs];
    }
    __syncthreads();
  }
  float mean = s1[0] * (1.0f / EE);
  float var = (s2[0] - (float)EE * mean * mean) * (1.0f / (EE - 1));  // ddof=1
  float std = sqrtf(fmaxf(var, 0.f));
#pragma unroll
  for (int t = 0; t < 4; ++t) {
    int e = threadIdx.x + t * 256;
    float div = std + 1e-6f + gshift[e];
    out[(long)row * EE + e] = gscale[e] * (x[t] - mean) / div;
  }
}

// ------------------------------------------------------------------- launcher
extern "C" void kernel_launch(void* const* d_in, const int* in_sizes, int n_in,
                              void* d_out, int out_size, void* d_ws, size_t ws_size,
                              hipStream_t stream) {
  const float* query = (const float*)d_in[0];
  const float* key   = (const float*)d_in[1];
  const float* value = (const float*)d_in[2];
  const float* resid = (const float*)d_in[3];
  const float* Wq = (const float*)d_in[4];
  const float* bq = (const float*)d_in[5];
  const float* Wk = (const float*)d_in[6];
  const float* bk = (const float*)d_in[7];
  const float* Wv = (const float*)d_in[8];
  const float* bv = (const float*)d_in[9];
  const float* gscale = (const float*)d_in[10];
  const float* gshift = (const float*)d_in[11];
  float* out = (float*)d_out;

  char* w = (char*)d_ws;
  size_t off = 0;
  auto take = [&](size_t bytes) -> char* {
    char* p = w + off;
    off += (bytes + 255) & ~(size_t)255;
    return p;
  };

  unsigned short* Xq  = (unsigned short*)take((size_t)NX * 2);
  unsigned short* Xk  = (unsigned short*)take((size_t)NX * 2);
  unsigned short* Xv  = (unsigned short*)take((size_t)NX * 2);
  unsigned short* WqT = (unsigned short*)take((size_t)NW * 2);
  unsigned short* WkT = (unsigned short*)take((size_t)NW * 2);
  unsigned short* WvT = (unsigned short*)take((size_t)NW * 2);
  unsigned short* Qb  = (unsigned short*)take((size_t)NBHS * DD * 2);
  unsigned short* Kb  = (unsigned short*)take((size_t)NBHS * DD * 2);
  unsigned short* Vt  = (unsigned short*)take((size_t)NBHS * DD * 2);
  float* qn   = (float*)take((size_t)NBHS * 4);
  float* kn   = (float*)take((size_t)NBHS * 4);
  float* aout = (float*)take((size_t)NX * 4);

  cvt_x_kernel<<<NX / 8 / 256, 256, 0, stream>>>(query, key, value, Xq, Xk, Xv);
  cvt_w_kernel<<<NW / 256, 256, 0, stream>>>(Wq, Wk, Wv, WqT, WkT, WvT);

  proj_gemm_kernel<<<512, 128, 0, stream>>>(Xq, WqT, bq, Qb, 0);
  proj_gemm_kernel<<<512, 128, 0, stream>>>(Xk, WkT, bk, Kb, 1);
  proj_gemm_kernel<<<512, 128, 0, stream>>>(Xv, WvT, bv, Vt, 2);

  rownorm_kernel<<<NBHS / 256, 256, 0, stream>>>(Qb, Kb, qn, kn);

  flash_attn_kernel<<<512, 128, 0, stream>>>(Qb, Kb, Vt, qn, kn, aout);

  resid_ln_kernel<<<NROWS, 256, 0, stream>>>(aout, resid, gscale, gshift, out);
}
